// SNNToy_13615046328449
// MI455X (gfx1250) — compile-verified, matrix operand order checked by reading, not run on hardware
//
#include <hip/hip_runtime.h>

typedef __attribute__((ext_vector_type(16))) _Float16 v16h;
typedef __attribute__((ext_vector_type(4)))  _Float16 v4h;
typedef __attribute__((ext_vector_type(8)))  float    v8f;
typedef __attribute__((ext_vector_type(4)))  float    v4f;

#define T_STEPS 50
#define BATCH   1024
#define NIN     784
#define HID     256
#define NLAB    10
#define KSTEPS  25          // K padded 784 -> 800 (25 x 32)

union Frag16 { v16h h; v4f f[2]; };

static __device__ __forceinline__ v4h cvt4(v4f v) {
  v4h r;
  r.x = (_Float16)v.x; r.y = (_Float16)v.y;
  r.z = (_Float16)v.z; r.w = (_Float16)v.w;
  return r;
}

// -------------------------------------------------------------------------
// Kernel 1: drive[T*B, H] = X[T*B, NIN] @ Wi.T  (f16 WMMA, f32 accumulate)
// Block: 256 threads = 8 waves, macro tile 64(M) x 256(N full width).
// Wave w owns tiles (tm = (w&1)*2 + {0,1}) x (tn = (w>>1)*4 + {0..3}) -> 8
// WMMAs per k-step. Double-buffered LDS; x is read from HBM exactly once.
// grid = 51200/64 = 800 blocks.
// -------------------------------------------------------------------------
__global__ __launch_bounds__(256) void snn_input_gemm(
    const float* __restrict__ x, const float* __restrict__ Wi,
    float* __restrict__ drive)
{
  __shared__ _Float16 As[2][64 * 32];   // [m][k]  (4 KB per buffer)
  __shared__ _Float16 Bs[2][256 * 32];  // [n][k]  (16 KB per buffer)

  const int m0   = blockIdx.x * 64;
  const int tid  = threadIdx.x;
  const int lane = tid & 31;
  const int wave = tid >> 5;
  const int tmb  = (wave & 1) * 2;    // first of 2 tm tiles
  const int tnb  = (wave >> 1) * 4;   // first of 4 tn tiles
  const int half = lane >> 4;
  const int lrow = lane & 15;

  // staging-register assignment
  const int rA = tid >> 2;            // 0..63   (A row)
  const int cA = (tid & 3) * 8;       // 0,8,16,24
  const int rB = tid;                 // 0..255  (B row = Wi row)

  const v4f vzero = {0.0f, 0.0f, 0.0f, 0.0f};
  v4f ra0, ra1, rb[8];

  // ---- branchless clamped loads for one k-tile into registers ----
  auto loadRegs = [&](int kk) {
    const int k0 = kk * 32;
    {
      const float* p = x + (size_t)(m0 + rA) * NIN;
      const int ka = k0 + cA;               // multiple of 8
      const bool ok = ka < NIN;             // ka<784 => ka+7 <= 783 valid
      const int kc = ok ? ka : 0;
      ra0 = *(const v4f*)(p + kc);
      ra1 = *(const v4f*)(p + kc + 4);
      if (!ok) { ra0 = vzero; ra1 = vzero; }
    }
    {
      const float* p = Wi + (size_t)rB * NIN;
      #pragma unroll
      for (int j = 0; j < 8; ++j) {
        const int kb = k0 + j * 4;          // multiple of 4
        const bool ok = kb < NIN;
        rb[j] = *(const v4f*)(p + (ok ? kb : 0));
        if (!ok) rb[j] = vzero;
      }
    }
  };

  auto storeRegs = [&](int buf) {
    _Float16* ad = &As[buf][rA * 32 + cA];
    *(v4h*)(ad)     = cvt4(ra0);
    *(v4h*)(ad + 4) = cvt4(ra1);
    _Float16* bd = &Bs[buf][rB * 32];
    #pragma unroll
    for (int j = 0; j < 8; ++j)
      *(v4h*)(bd + j * 4) = cvt4(rb[j]);
  };

  v8f acc[2][4] = {};

  loadRegs(0);
  for (int kk = 0; kk < KSTEPS; ++kk) {
    const int buf = kk & 1;
    storeRegs(buf);
    __syncthreads();
    if (kk + 1 < KSTEPS) loadRegs(kk + 1);   // overlaps with WMMA below

    Frag16 fa[2], fb[4];
    #pragma unroll
    for (int c2 = 0; c2 < 2; ++c2) {
      const _Float16* ap = &As[buf][((tmb + c2) * 16 + lrow) * 32 + half * 8];
      fa[c2].f[0] = *(const v4f*)(ap);
      fa[c2].f[1] = *(const v4f*)(ap + 16);
    }
    #pragma unroll
    for (int c = 0; c < 4; ++c) {
      const _Float16* bp = &Bs[buf][((tnb + c) * 16 + lrow) * 32 + half * 8];
      fb[c].f[0] = *(const v4f*)(bp);
      fb[c].f[1] = *(const v4f*)(bp + 16);
    }
    #pragma unroll
    for (int c2 = 0; c2 < 2; ++c2)
      #pragma unroll
      for (int c = 0; c < 4; ++c)
        acc[c2][c] = __builtin_amdgcn_wmma_f32_16x16x32_f16(
            false, fa[c2].h, false, fb[c].h, (short)0, acc[c2][c],
            false, false);
    __syncthreads();
  }

  #pragma unroll
  for (int c2 = 0; c2 < 2; ++c2) {
    #pragma unroll
    for (int c = 0; c < 4; ++c) {
      const int n = (tnb + c) * 16 + lrow;
      #pragma unroll
      for (int r = 0; r < 8; ++r) {
        const int m = m0 + (tmb + c2) * 16 + r + 8 * half;
        drive[(size_t)m * HID + n] = acc[c2][c][r];
      }
    }
  }
}

// -------------------------------------------------------------------------
// Kernel 2: sequential LIF scan. Each block owns 16 batch rows for all 50
// steps; recurrent GEMM z @ Wr.T done with WMMA, Wr(f16) resident in LDS.
// Block: 128 threads = 4 waves; wave w covers hidden tiles tn = 4w..4w+3.
// grid = 1024/16 = 64 blocks. Accumulates spike counts S[b,h].
// -------------------------------------------------------------------------
__global__ __launch_bounds__(128) void snn_scan(
    const float* __restrict__ drive, const float* __restrict__ Wr,
    float* __restrict__ S)
{
  extern __shared__ char smem[];
  _Float16* wr = (_Float16*)smem;                   // [256][256] f16 (128 KB)
  _Float16* zl = (_Float16*)(smem + HID * HID * 2); // [16][256]  f16 (8 KB)

  const int tid  = threadIdx.x;
  const int lane = tid & 31;
  const int wave = tid >> 5;          // 0..3
  const int b0   = blockIdx.x * 16;
  const int tnb  = wave * 4;          // first hidden tile of this wave
  const int half = lane >> 4;
  const int lrow = lane & 15;
  const int brow = 8 * half;          // + r gives local batch row

  // stage Wr -> f16 LDS (row-major [h][j]: contiguous K per output column)
  {
    const v4f* src = (const v4f*)Wr;
    for (int idx = tid; idx < (HID * HID) / 4; idx += 128)
      *(v4h*)(wr + idx * 4) = cvt4(src[idx]);
  }
  for (int idx = tid; idx < (16 * HID) / 4; idx += 128) {  // z0 = 0
    v4h z = {(_Float16)0.f, (_Float16)0.f, (_Float16)0.f, (_Float16)0.f};
    *(v4h*)(zl + idx * 4) = z;
  }
  __syncthreads();

  v8f vmem[4] = {{0}}, cur[4] = {{0}}, scnt[4] = {{0}};

  for (int t = 0; t < T_STEPS; ++t) {
    // load drive[t] for this thread's C-layout elements; prefetch t+1
    v8f drv[4];
    #pragma unroll
    for (int c = 0; c < 4; ++c) {
      const int h = (tnb + c) * 16 + lrow;
      const float* dp = drive + ((size_t)t * BATCH + b0 + brow) * HID + h;
      #pragma unroll
      for (int r = 0; r < 8; ++r) drv[c][r] = dp[(size_t)r * HID];
      if (t + 1 < T_STEPS)
        __builtin_prefetch(dp + (size_t)BATCH * HID, 0, 0);
    }

    // rec = z_{t-1} @ Wr.T  (K = 256 -> 8 WMMA steps, A frag reused x4)
    v8f rec[4] = {{0}};
    #pragma unroll
    for (int kk = 0; kk < 8; ++kk) {
      Frag16 fa;
      const _Float16* ap = zl + lrow * HID + kk * 32 + half * 8;
      fa.f[0] = *(const v4f*)(ap);
      fa.f[1] = *(const v4f*)(ap + 16);
      #pragma unroll
      for (int c = 0; c < 4; ++c) {
        Frag16 fb;
        const _Float16* bp =
            wr + ((tnb + c) * 16 + lrow) * HID + kk * 32 + half * 8;
        fb.f[0] = *(const v4f*)(bp);
        fb.f[1] = *(const v4f*)(bp + 16);
        rec[c] = __builtin_amdgcn_wmma_f32_16x16x32_f16(
            false, fa.h, false, fb.h, (short)0, rec[c], false, false);
      }
    }
    __syncthreads();  // all waves done reading z_{t-1}

    // LIF elementwise update; publish z_t to LDS for next step
    #pragma unroll
    for (int c = 0; c < 4; ++c) {
      const int h = (tnb + c) * 16 + lrow;
      #pragma unroll
      for (int r = 0; r < 8; ++r) {
        const float v  = vmem[c][r];
        const float i  = cur[c][r];
        const float vd = 0.95f * v + 0.05f * i;        // v + dt/tau_m*(i - v)
        const float zn = (vd > 0.5f) ? 1.0f : 0.0f;    // heaviside(v - v_th)
        vmem[c][r] = zn != 0.0f ? 0.0f : vd;           // reset to 0
        cur[c][r]  = 0.9f * i + drv[c][r] + rec[c][r]; // decay + drive + rec
        scnt[c][r] += zn;
        zl[(brow + r) * HID + h] = (_Float16)zn;
      }
    }
    __syncthreads();  // z_t visible to all waves
  }

  #pragma unroll
  for (int c = 0; c < 4; ++c) {
    const int h = (tnb + c) * 16 + lrow;
    #pragma unroll
    for (int r = 0; r < 8; ++r)
      S[(size_t)(b0 + brow + r) * HID + h] = scnt[c][r];
  }
}

// -------------------------------------------------------------------------
// Kernel 3: out[b,l] = dot(S[b,:], Wout[l,:]) / T + bout[l]
// (uses Σ_t z_t @ Wout.T = (Σ_t z_t) @ Wout.T; per-step bout sums to T*bout)
// -------------------------------------------------------------------------
__global__ __launch_bounds__(256) void snn_readout(
    const float* __restrict__ S, const float* __restrict__ Wout,
    const float* __restrict__ bout, float* __restrict__ out)
{
  const int idx = blockIdx.x * blockDim.x + threadIdx.x;
  if (idx >= BATCH * NLAB) return;
  const int b = idx / NLAB, l = idx % NLAB;
  const v4f* sp = (const v4f*)(S + (size_t)b * HID);
  const v4f* wp = (const v4f*)(Wout + (size_t)l * HID);
  float acc = 0.0f;
  #pragma unroll 4
  for (int h = 0; h < HID / 4; ++h) {
    const v4f s = sp[h], w = wp[h];
    acc += s.x * w.x + s.y * w.y + s.z * w.z + s.w * w.w;
  }
  out[idx] = acc * (1.0f / T_STEPS) + bout[l];
}

extern "C" void kernel_launch(void* const* d_in, const int* in_sizes, int n_in,
                              void* d_out, int out_size, void* d_ws, size_t ws_size,
                              hipStream_t stream) {
  const float* x    = (const float*)d_in[0];
  const float* Wi   = (const float*)d_in[1];
  const float* Wr   = (const float*)d_in[2];
  const float* Wout = (const float*)d_in[3];
  const float* bout = (const float*)d_in[4];
  float* out = (float*)d_out;

  float* drive = (float*)d_ws;                              // 50*1024*256 f32
  float* S     = drive + (size_t)T_STEPS * BATCH * HID;     // 1024*256 f32

  snn_input_gemm<<<(T_STEPS * BATCH) / 64, 256, 0, stream>>>(x, Wi, drive);

  const size_t smem2 = (size_t)(HID * HID + 16 * HID) * sizeof(_Float16); // 136 KB
  snn_scan<<<BATCH / 16, 128, smem2, stream>>>(drive, Wr, S);

  snn_readout<<<(BATCH * NLAB + 255) / 256, 256, 0, stream>>>(S, Wout, bout, out);
}
